// ECC_Transformer_31756988187206
// MI455X (gfx1250) — matrix operand order repaired and checked by reading, not compiled, not run on hardware
//
#include <hip/hip_runtime.h>
#include <math.h>

// ---------------- problem constants ----------------
#define BB   16
#define SS   512
#define DD   512
#define HH   8
#define DK   64
#define NL   6
#define DFF  2048
#define NTOK (BB * SS)   // 8192

typedef __attribute__((ext_vector_type(16))) __bf16 bf16x16;
typedef __attribute__((ext_vector_type(8)))  __bf16 bf16x8;
typedef __attribute__((ext_vector_type(8)))  float  f32x8;

// ---------------- CDNA5 async global->LDS helpers ----------------
__device__ __forceinline__ void async_load_b128(unsigned int lds_off, const void* gaddr)
{
    // GLOBAL_LOAD_ASYNC_TO_LDS_B128: VDST = LDS byte offset (VGPR), VADDR = 64-bit global
    asm volatile("global_load_async_to_lds_b128 %0, %1, off"
                 :: "v"(lds_off), "v"(gaddr)
                 : "memory");
}
#define ASYNC_WAIT_ALL()  asm volatile("s_wait_asynccnt 0x0" ::: "memory")
#define ASYNC_WAIT_4()    asm volatile("s_wait_asynccnt 0x4" ::: "memory")

__device__ __forceinline__ unsigned int lds_offset_of(const void* p)
{
    // addrspace(3) -> generic keeps the LDS offset in the low 32 bits
    return (unsigned int)(unsigned long long)p;
}

// =====================================================================
// f32 -> bf16 bulk convert (weights, once per launch)
// =====================================================================
__global__ __launch_bounds__(256) void cvt_kernel(const float* __restrict__ in,
                                                  __bf16* __restrict__ out,
                                                  unsigned long long n)
{
    unsigned long long i = ((unsigned long long)blockIdx.x * 256 + threadIdx.x) * 4;
    #pragma unroll
    for (int t = 0; t < 4; ++t)
        if (i + t < n) out[i + t] = (__bf16)in[i + t];
}

// =====================================================================
// Positional encoding table: pe[s][d]  (s==0 -> cls_pe)
// =====================================================================
__global__ __launch_bounds__(256) void pe_kernel(const float* __restrict__ coords,
                                                 const float* __restrict__ cls_pe,
                                                 float* __restrict__ pe)
{
    int idx = blockIdx.x * 256 + threadIdx.x;
    if (idx >= SS * DD) return;
    int d = idx % DD;
    int s = idx / DD;
    float val;
    if (s == 0) {
        val = cls_pe[d];
    } else {
        const int half = DD / 2;                 // 256
        int dd = d % half;
        float c = coords[(s - 1) * 2 + (d >= half ? 1 : 0)];
        int j = dd >> 1;
        float div = expf((float)(2 * j) * (-9.210340371976184f / (float)half));
        float ang = c * div;
        val = (dd & 1) ? cosf(ang) : sinf(ang);
    }
    pe[idx] = val;
}

__global__ __launch_bounds__(256) void addpe_kernel(const float* __restrict__ x,
                                                    const float* __restrict__ pe,
                                                    float* __restrict__ xw)
{
    size_t idx = (size_t)blockIdx.x * 256 + threadIdx.x;
    if (idx >= (size_t)NTOK * DD) return;
    size_t sd = idx % ((size_t)SS * DD);
    xw[idx] = x[idx] + pe[sd];
}

// =====================================================================
// Relative position bias: bias[h][i][j], CLS row/col (i==0 || j==0) == 0
// =====================================================================
__global__ __launch_bounds__(256) void bias_kernel(const float* __restrict__ coords,
                                                   const float* __restrict__ emb,
                                                   float* __restrict__ bias)
{
    size_t idx = (size_t)blockIdx.x * 256 + threadIdx.x;
    if (idx >= (size_t)HH * SS * SS) return;
    int j = (int)(idx % SS);
    int i = (int)((idx / SS) % SS);
    int h = (int)(idx / ((size_t)SS * SS));
    float v = 0.f;
    if (i > 0 && j > 0) {
        float dx = coords[(i - 1) * 2 + 0] - coords[(j - 1) * 2 + 0];
        float dy = coords[(i - 1) * 2 + 1] - coords[(j - 1) * 2 + 1];
        float dist = sqrtf(dx * dx + dy * dy);
        float t = fminf(fmaxf(dist / 10.0f, 0.f), 1.f) * 31.0f;
        int bk = (int)t;
        v = emb[bk * HH + h];
    }
    bias[idx] = v;
}

// =====================================================================
// Row LayerNorm (rows of 512). OUT_BF16 selects output type.
// =====================================================================
template <bool OUT_BF16>
__global__ __launch_bounds__(256) void ln_kernel(const float* __restrict__ in,
                                                 const float* __restrict__ g,
                                                 const float* __restrict__ b,
                                                 float* __restrict__ outf,
                                                 __bf16* __restrict__ outb)
{
    __shared__ float red[256];
    const int row = blockIdx.x;
    const int tid = threadIdx.x;
    const float* xr = in + (size_t)row * DD;

    float s = 0.f;
    for (int i = tid; i < DD; i += 256) s += xr[i];
    red[tid] = s; __syncthreads();
    for (int off = 128; off > 0; off >>= 1) {
        if (tid < off) red[tid] += red[tid + off];
        __syncthreads();
    }
    float mu = red[0] / (float)DD;
    __syncthreads();

    float v = 0.f;
    for (int i = tid; i < DD; i += 256) { float d = xr[i] - mu; v += d * d; }
    red[tid] = v; __syncthreads();
    for (int off = 128; off > 0; off >>= 1) {
        if (tid < off) red[tid] += red[tid + off];
        __syncthreads();
    }
    float rstd = rsqrtf(red[0] / (float)DD + 1e-5f);
    __syncthreads();

    for (int i = tid; i < DD; i += 256) {
        float val = (xr[i] - mu) * rstd * g[i] + b[i];
        if (OUT_BF16) outb[(size_t)row * DD + i] = (__bf16)val;
        else          outf[(size_t)row * DD + i] = val;
    }
}

// =====================================================================
// WMMA GEMM (bf16 x bf16 -> f32): C = act(A @ W + bias) (+ residual)
// Block tile 128x128, 8 waves (4x2), wave tile 32x64 = 2x4 WMMA 16x16x32.
// Tiles staged via GLOBAL_LOAD_ASYNC_TO_LDS_B128, double-buffered.
// =====================================================================
__device__ __forceinline__ void gemm_issue(const __bf16* __restrict__ A,
                                           const __bf16* __restrict__ W,
                                           int K, int N, int mblock, int nblock, int k0,
                                           unsigned int asA, unsigned int asB,
                                           int atr, int atc, int btr, int btc)
{
    const __bf16* ga = A + (size_t)(mblock + atr) * K + (k0 + atc);
    async_load_b128(asA + (unsigned)(atr * 32 + atc) * 2u, ga);
    async_load_b128(asA + (unsigned)(atr * 32 + atc + 8) * 2u, ga + 8);
    const __bf16* gb = W + (size_t)(k0 + btr) * N + (nblock + btc);
    async_load_b128(asB + (unsigned)(btr * 128 + btc) * 2u, gb);
    async_load_b128(asB + (unsigned)(btr * 128 + btc + 8) * 2u, gb + 8);
    if (k0 + 64 < K) {   // prefetch 2 k-steps ahead
        __builtin_prefetch(ga + 64, 0, 0);
        __builtin_prefetch(gb + (size_t)64 * N, 0, 0);
    }
}

__global__ __launch_bounds__(256) void gemm_kernel(const __bf16* __restrict__ A,
                                                   const __bf16* __restrict__ W,
                                                   const float* __restrict__ bias,
                                                   const float* __restrict__ resid,
                                                   float* __restrict__ Cf,
                                                   __bf16* __restrict__ Cb,
                                                   int M, int N, int K, int act)
{
    __shared__ alignas(32) __bf16 As[2][128 * 32];   // 16 KB
    __shared__ alignas(32) __bf16 Bs[2][32 * 128];   // 16 KB

    const int tid  = threadIdx.x;
    const int lane = tid & 31;
    const int wave = tid >> 5;
    const int wr = wave >> 1;            // 0..3  (32-row slab)
    const int wc = wave & 1;             // 0..1  (64-col slab)
    const int mblock = blockIdx.y * 128;
    const int nblock = blockIdx.x * 128;
    const int ml  = lane & 15;
    const int klo = (lane >> 4) << 3;    // 0 or 8
    const int mhi = (lane >> 4) << 3;    // C-row offset 0 or 8

    const int atr = tid >> 1, atc = (tid & 1) * 16;   // A: 2 thr/row, 32B each
    const int btr = tid >> 3, btc = (tid & 7) * 16;   // B: 8 thr/row, 32B each

    const unsigned int asA0 = lds_offset_of(&As[0][0]);
    const unsigned int asA1 = lds_offset_of(&As[1][0]);
    const unsigned int asB0 = lds_offset_of(&Bs[0][0]);
    const unsigned int asB1 = lds_offset_of(&Bs[1][0]);

    f32x8 vzero = {};
    f32x8 acc[2][4];
    #pragma unroll
    for (int mi = 0; mi < 2; ++mi)
        #pragma unroll
        for (int ni = 0; ni < 4; ++ni) acc[mi][ni] = vzero;

    gemm_issue(A, W, K, N, mblock, nblock, 0, asA0, asB0, atr, atc, btr, btc);

    const int nk = K / 32;
    for (int kt = 0; kt < nk; ++kt) {
        const int p = kt & 1;
        if (kt + 1 < nk) {
            gemm_issue(A, W, K, N, mblock, nblock, (kt + 1) * 32,
                       p ? asA0 : asA1, p ? asB0 : asB1, atr, atc, btr, btc);
            ASYNC_WAIT_4();
        } else {
            ASYNC_WAIT_ALL();
        }
        __syncthreads();

        bf16x16 af[2], bf[4];
        #pragma unroll
        for (int mi = 0; mi < 2; ++mi) {
            const __bf16* pa = &As[p][(wr * 32 + mi * 16 + ml) * 32];
            bf16x8 lo = *(const bf16x8*)(pa + klo);
            bf16x8 hi = *(const bf16x8*)(pa + 16 + klo);
            af[mi] = __builtin_shufflevector(lo, hi, 0,1,2,3,4,5,6,7,8,9,10,11,12,13,14,15);
        }
        #pragma unroll
        for (int ni = 0; ni < 4; ++ni)
            #pragma unroll
            for (int e = 0; e < 8; ++e) {
                bf[ni][e]     = Bs[p][(klo + e) * 128 + wc * 64 + ni * 16 + ml];
                bf[ni][e + 8] = Bs[p][(16 + klo + e) * 128 + wc * 64 + ni * 16 + ml];
            }

        #pragma unroll
        for (int mi = 0; mi < 2; ++mi)
            #pragma unroll
            for (int ni = 0; ni < 4; ++ni)
                acc[mi][ni] = __builtin_amdgcn_wmma_f32_16x16x32_bf16(
                    false, af[mi], false, bf[ni], (short)0, acc[mi][ni], false, false);
        __syncthreads();
    }

    #pragma unroll
    for (int mi = 0; mi < 2; ++mi) {
        #pragma unroll
        for (int ni = 0; ni < 4; ++ni) {
            const int col = nblock + wc * 64 + ni * 16 + ml;
            #pragma unroll
            for (int r = 0; r < 8; ++r) {
                const int row = mblock + wr * 32 + mi * 16 + r + mhi;
                float val = acc[mi][ni][r] + bias[col];
                if (act == 1)   // exact GELU
                    val = 0.5f * val * (1.0f + erff(val * 0.70710678118654752f));
                if (Cf) {
                    if (resid) val += resid[(size_t)row * N + col];
                    Cf[(size_t)row * N + col] = val;
                } else {
                    Cb[(size_t)row * N + col] = (__bf16)val;
                }
            }
        }
    }
}

// =====================================================================
// Fused attention: per (b, h, 32 query rows). 64 threads = 2 waves,
// each wave owns 16 q-rows. Scores (32x512 f32) live in LDS.
// Q/K/V are bf16 [B,S,D] with head-major columns; O written as bf16.
// =====================================================================
__global__ __launch_bounds__(64) void attn_kernel(const __bf16* __restrict__ Q,
                                                  const __bf16* __restrict__ Km,
                                                  const __bf16* __restrict__ Vm,
                                                  const float* __restrict__ bias,
                                                  const unsigned char* __restrict__ mask,
                                                  __bf16* __restrict__ O)
{
    __shared__ float sc[32 * SS];   // 64 KB
    const int b  = blockIdx.z;
    const int h  = blockIdx.y;
    const int q0 = blockIdx.x * 32;
    const int tid  = threadIdx.x;
    const int lane = tid & 31;
    const int wave = tid >> 5;
    const int ml  = lane & 15;
    const int klo = (lane >> 4) << 3;
    const int mhi = (lane >> 4) << 3;
    const float scale = 0.125f;   // 1/sqrt(64)

    // ---- Phase 1: scores = Q K^T * scale + bias (mask) ----
    bf16x16 afr[2];
    {
        const int qrow = q0 + wave * 16 + ml;
        const __bf16* qp = Q + ((size_t)(b * SS + qrow) * DD + h * DK);
        #pragma unroll
        for (int kc = 0; kc < 2; ++kc) {
            bf16x8 lo = *(const bf16x8*)(qp + kc * 32 + klo);
            bf16x8 hi = *(const bf16x8*)(qp + kc * 32 + 16 + klo);
            afr[kc] = __builtin_shufflevector(lo, hi, 0,1,2,3,4,5,6,7,8,9,10,11,12,13,14,15);
        }
    }
    const __bf16* kbase = Km + (size_t)(b * SS) * DD + h * DK;
    for (int nt = 0; nt < SS / 16; ++nt) {
        const int n0 = nt * 16;
        f32x8 acc = {};
        #pragma unroll
        for (int kc = 0; kc < 2; ++kc) {
            bf16x16 bf;
            const __bf16* kp = kbase + (size_t)(n0 + ml) * DD + kc * 32;   // ml = key column
            bf16x8 lo = *(const bf16x8*)(kp + klo);
            bf16x8 hi = *(const bf16x8*)(kp + 16 + klo);
            bf = __builtin_shufflevector(lo, hi, 0,1,2,3,4,5,6,7,8,9,10,11,12,13,14,15);
            acc = __builtin_amdgcn_wmma_f32_16x16x32_bf16(
                false, afr[kc], false, bf, (short)0, acc, false, false);
        }
        const int col = n0 + ml;
        #pragma unroll
        for (int r = 0; r < 8; ++r) {
            int lrow = wave * 16 + r + mhi;             // 0..31 within block
            float s = acc[r] * scale
                    + bias[((size_t)h * SS + (q0 + lrow)) * SS + col];
            if (mask[b * SS + col]) s = -1e9f;
            sc[lrow * SS + col] = s;
        }
    }
    __syncthreads();

    // ---- softmax (one thread per row) ----
    if (tid < 32) {
        float* row = sc + tid * SS;
        float m = -3.4e38f;
        for (int j = 0; j < SS; ++j) m = fmaxf(m, row[j]);
        float ssum = 0.f;
        for (int j = 0; j < SS; ++j) { float e = expf(row[j] - m); row[j] = e; ssum += e; }
        float inv = 1.f / ssum;
        for (int j = 0; j < SS; ++j) row[j] *= inv;
    }
    __syncthreads();

    // ---- Phase 2: O = P V ----
    f32x8 vzero = {};
    f32x8 oacc[4];
    oacc[0] = vzero; oacc[1] = vzero; oacc[2] = vzero; oacc[3] = vzero;
    const __bf16* vbase = Vm + (size_t)(b * SS) * DD + h * DK;
    for (int kc = 0; kc < SS / 32; ++kc) {
        bf16x16 a;
        const float* prow = sc + (wave * 16 + ml) * SS + kc * 32;
        #pragma unroll
        for (int e = 0; e < 8; ++e) {
            a[e]     = (__bf16)prow[klo + e];
            a[e + 8] = (__bf16)prow[16 + klo + e];
        }
        const __bf16* vp = vbase + (size_t)(kc * 32) * DD + ml;  // ml = output column
        #pragma unroll
        for (int ni = 0; ni < 4; ++ni) {
            bf16x16 bf;
            #pragma unroll
            for (int e = 0; e < 8; ++e) {
                bf[e]     = vp[(size_t)(klo + e) * DD + ni * 16];
                bf[e + 8] = vp[(size_t)(16 + klo + e) * DD + ni * 16];
            }
            oacc[ni] = __builtin_amdgcn_wmma_f32_16x16x32_bf16(
                false, a, false, bf, (short)0, oacc[ni], false, false);
        }
    }
    #pragma unroll
    for (int ni = 0; ni < 4; ++ni) {
        #pragma unroll
        for (int r = 0; r < 8; ++r) {
            int m_abs = q0 + wave * 16 + r + mhi;
            O[((size_t)(b * SS + m_abs)) * DD + h * DK + ni * 16 + ml] = (__bf16)oacc[ni][r];
        }
    }
}

// =====================================================================
// Host-side launch sequence
// =====================================================================
extern "C" void kernel_launch(void* const* d_in, const int* in_sizes, int n_in,
                              void* d_out, int out_size, void* d_ws, size_t ws_size,
                              hipStream_t stream)
{
    (void)in_sizes; (void)n_in; (void)out_size; (void)ws_size;

    const float* x        = (const float*)d_in[0];
    const float* coords   = (const float*)d_in[1];
    const float* cls_pe   = (const float*)d_in[2];
    const float* bias_emb = (const float*)d_in[3];
    const float* attn_w   = (const float*)d_in[4];   // [NL,4,D,D]
    const float* attn_b   = (const float*)d_in[5];   // [NL,4,D]
    const float* ff_w1    = (const float*)d_in[6];   // [NL,D,DFF]
    const float* ff_b1    = (const float*)d_in[7];   // [NL,DFF]
    const float* ff_w2    = (const float*)d_in[8];   // [NL,DFF,D]
    const float* ff_b2    = (const float*)d_in[9];   // [NL,D]
    const float* ln_g     = (const float*)d_in[10];  // [NL,2,D]
    const float* ln_b     = (const float*)d_in[11];
    const float* norm_g   = (const float*)d_in[12];
    const float* norm_b   = (const float*)d_in[13];
    const float* norm2_g  = (const float*)d_in[14];
    const float* norm2_b  = (const float*)d_in[15];
    const unsigned char* mask = (const unsigned char*)d_in[16];

    // ---- workspace layout ----
    const size_t PE_SZ   = (size_t)SS * DD;            // 262144 f32
    const size_t BIAS_SZ = (size_t)HH * SS * SS;       // 2097152 f32
    const size_t ACT_SZ  = (size_t)NTOK * DD;          // 4194304 elems
    const size_t FF_SZ   = (size_t)NTOK * DFF;         // 16777216 elems
    const size_t WATTN_SZ = (size_t)NL * 4 * DD * DD;  // 6291456
    const size_t WFF_SZ   = (size_t)NL * DD * DFF;     // 6291456

    float* pe   = (float*)d_ws;
    float* bias = pe + PE_SZ;
    float* xw   = bias + BIAS_SZ;
    __bf16* w_attn = (__bf16*)(xw + ACT_SZ);
    __bf16* w_ff1  = w_attn + WATTN_SZ;
    __bf16* w_ff2  = w_ff1 + WFF_SZ;
    __bf16* xln = w_ff2 + WFF_SZ;
    __bf16* q   = xln + ACT_SZ;
    __bf16* k   = q + ACT_SZ;
    __bf16* v   = k + ACT_SZ;
    __bf16* o   = v + ACT_SZ;
    __bf16* ffh = o + ACT_SZ;
    (void)FF_SZ;

    // ---- one-time prep ----
    cvt_kernel<<<(unsigned)((WATTN_SZ / 4 + 255) / 256), 256, 0, stream>>>(attn_w, w_attn, WATTN_SZ);
    cvt_kernel<<<(unsigned)((WFF_SZ   / 4 + 255) / 256), 256, 0, stream>>>(ff_w1,  w_ff1,  WFF_SZ);
    cvt_kernel<<<(unsigned)((WFF_SZ   / 4 + 255) / 256), 256, 0, stream>>>(ff_w2,  w_ff2,  WFF_SZ);
    pe_kernel  <<<(SS * DD + 255) / 256, 256, 0, stream>>>(coords, cls_pe, pe);
    bias_kernel<<<(unsigned)((BIAS_SZ + 255) / 256), 256, 0, stream>>>(coords, bias_emb, bias);
    addpe_kernel<<<(unsigned)((ACT_SZ + 255) / 256), 256, 0, stream>>>(x, pe, xw);

    dim3 g512(DD / 128, NTOK / 128);   // (4, 64)
    dim3 gff (DFF / 128, NTOK / 128);  // (16, 64)
    dim3 gattn(SS / 32, HH, BB);       // (16, 8, 16)

    for (int i = 0; i < NL; ++i) {
        const __bf16* Wq = w_attn + ((size_t)i * 4 + 0) * DD * DD;
        const __bf16* Wk = w_attn + ((size_t)i * 4 + 1) * DD * DD;
        const __bf16* Wv = w_attn + ((size_t)i * 4 + 2) * DD * DD;
        const __bf16* Wo = w_attn + ((size_t)i * 4 + 3) * DD * DD;
        const float* bq = attn_b + ((size_t)i * 4 + 0) * DD;
        const float* bk = attn_b + ((size_t)i * 4 + 1) * DD;
        const float* bv = attn_b + ((size_t)i * 4 + 2) * DD;
        const float* bo = attn_b + ((size_t)i * 4 + 3) * DD;

        ln_kernel<true><<<NTOK, 256, 0, stream>>>(xw, ln_g + ((size_t)i * 2 + 0) * DD,
                                                  ln_b + ((size_t)i * 2 + 0) * DD, nullptr, xln);
        gemm_kernel<<<g512, 256, 0, stream>>>(xln, Wq, bq, nullptr, nullptr, q, NTOK, DD, DD, 0);
        gemm_kernel<<<g512, 256, 0, stream>>>(xln, Wk, bk, nullptr, nullptr, k, NTOK, DD, DD, 0);
        gemm_kernel<<<g512, 256, 0, stream>>>(xln, Wv, bv, nullptr, nullptr, v, NTOK, DD, DD, 0);

        attn_kernel<<<gattn, 64, 0, stream>>>(q, k, v, bias, mask, o);

        gemm_kernel<<<g512, 256, 0, stream>>>(o, Wo, bo, xw, xw, nullptr, NTOK, DD, DD, 0);

        ln_kernel<true><<<NTOK, 256, 0, stream>>>(xw, ln_g + ((size_t)i * 2 + 1) * DD,
                                                  ln_b + ((size_t)i * 2 + 1) * DD, nullptr, xln);
        gemm_kernel<<<gff, 256, 0, stream>>>(xln, w_ff1 + (size_t)i * DD * DFF,
                                             ff_b1 + (size_t)i * DFF, nullptr, nullptr, ffh,
                                             NTOK, DFF, DD, 1);
        gemm_kernel<<<g512, 256, 0, stream>>>(ffh, w_ff2 + (size_t)i * DFF * DD,
                                              ff_b2 + (size_t)i * DD, xw, xw, nullptr,
                                              NTOK, DD, DFF, 0);
        if (i == NL / 2 - 1)   // after layer 3 of 6
            ln_kernel<false><<<NTOK, 256, 0, stream>>>(xw, norm2_g, norm2_b, xw, nullptr);
    }
    ln_kernel<false><<<NTOK, 256, 0, stream>>>(xw, norm_g, norm_b, (float*)d_out, nullptr);
}
